// LabHeadModule_17841294147649
// MI455X (gfx1250) — matrix operand order repaired and checked by reading, not compiled, not run on hardware
//
#include <hip/hip_runtime.h>

// ---------------------------------------------------------------------------
// Problem constants (from reference)
// ---------------------------------------------------------------------------
enum { B_ = 4, S_ = 2048, H_ = 1024, NE_ = 4, NC_ = 16, CD_ = 64, OD_ = 128 };
#define TAU_   0.5f
#define TEMP_  0.07f
#define EPS_   1e-5f

typedef __bf16 bf16_t;
typedef __attribute__((ext_vector_type(16))) __bf16 v16bf;
typedef __attribute__((ext_vector_type(8)))  __bf16 v8bf;
typedef __attribute__((ext_vector_type(8)))  float  v8f;
typedef __attribute__((ext_vector_type(4)))  unsigned int v4u;
typedef __attribute__((ext_vector_type(8)))  int v8i;
typedef __attribute__((ext_vector_type(4)))  int v4i;

__device__ __forceinline__ float gelu_f(float x) {
    return 0.5f * x * (1.0f + erff(x * 0.70710678118654752f));
}

// ---------------------------------------------------------------------------
// TDM: one descriptor moves a rows x 64 bf16 tile (row-major, strideElems
// between rows) from global into LDS, padding each 128B row with 16B so the
// LDS row stride is 144B (72 bf16) to match the fragment loader.
// D# layout per CDNA5 ISA.
// Toolchain uses the 6-arg builtin: (g0, g1, g2, g3, g4(int32x8), cpol).
// ---------------------------------------------------------------------------
__device__ __forceinline__ void tdm_load_tile(
    unsigned int ldsByte, const void* gsrc, int rows, long long strideElems)
{
    unsigned long long ga = (unsigned long long)(size_t)gsrc;
    v4u g0;
    g0[0] = 1u;                                    // count=1 (valid), user mode
    g0[1] = ldsByte;                               // lds_addr [63:32]
    g0[2] = (unsigned int)ga;                      // global_addr[31:0]
    g0[3] = (unsigned int)((ga >> 32) & 0x01ffffffu)  // global_addr[56:32]
          | 0x80000000u;                           // type=2 ("image") [127:126]
    v8i g1;
    g1[0] = (int)((1u << 16)       // data_size = 2 bytes
                | (1u << 20)       // pad_enable
                | (4u << 22)       // pad_interval: 32 dwords = 128B
                | (3u << 25));     // pad_amount: 4 dwords = 16B
    g1[1] = (int)((64u & 0xffffu) << 16);              // tensor_dim0=64 (lo16)
    g1[2] = (int)(((unsigned)rows & 0xffffu) << 16);   // dim0 hi=0 | tensor_dim1 lo16
    g1[3] = (int)(64u << 16);                          // dim1 hi=0 | tile_dim0=64
    g1[4] = (int)((unsigned)rows & 0xffffu);           // tile_dim1=rows, tile_dim2=0
    g1[5] = (int)(unsigned int)((unsigned long long)strideElems);        // stride0 lo32
    g1[6] = (int)(unsigned int)(((unsigned long long)strideElems >> 32) & 0xffffu);
    g1[7] = 0;                                         // dim1_stride (unused, 2D)
    v4i z4 = {0, 0, 0, 0};
    v8i z8 = {0, 0, 0, 0, 0, 0, 0, 0};
    __builtin_amdgcn_tensor_load_to_lds(g0, g1, z4, z4, z8, 0);
}

// ---------------------------------------------------------------------------
// Batched GEMM:  C[z] = act(scale * A[z] @ B[z]^T + bias) (+ resid)
//   A : M x K row-major (lda)
//   Bt: N x K row-major (ldb)   <- "transB" storage, K contiguous
// bf16 inputs, f32 accumulation via v_wmma_f32_16x16x32_bf16.
//
// Block tile 128x64, 8 waves, each wave owns a 32x32 patch (2x2 WMMA tiles).
// K staged 64 at a time, double-buffered LDS:
//   - A tile (128x64): one TENSOR_LOAD_TO_LDS issued by wave 0
//   - B tile (64x64):  GLOBAL_LOAD_ASYNC_TO_LDS_B128 (2/thread)
// pipelined with s_wait_tensorcnt / s_wait_asynccnt; last stage peeled.
// Requires: M%128==0, N%64==0, K%64==0, lda/ldb multiples of 8.
// ---------------------------------------------------------------------------
#define AROWS 128
#define BROWS 64
#define LPAD  72   // LDS row stride in bf16 (64 + 8 pad; 144B, 16B-aligned)

__global__ void __launch_bounds__(256) gemm_wmma(
    const bf16_t* __restrict__ A, const bf16_t* __restrict__ Bt,
    const float* __restrict__ bias, const float* __restrict__ resid,
    float* __restrict__ outF, bf16_t* __restrict__ outB,
    int M, int N, int K, int lda, int ldb, int ldc,
    long long sA, long long sB, long long sC,
    float scale, int actGelu, int transOutB)
{
    __shared__ alignas(16) bf16_t As[2 * AROWS * LPAD];
    __shared__ alignas(16) bf16_t Bs[2 * BROWS * LPAD];

    const int tid  = threadIdx.x;
    const int wave = tid >> 5;
    const int lane = tid & 31;
    const int wm   = wave & 3;        // row group (32 rows each)
    const int wn   = wave >> 2;       // col group (32 cols each)
    const int lr   = lane & 15;
    const int half = lane >> 4;
    const int m0   = blockIdx.y * 128;
    const int n0   = blockIdx.x * 64;

    const bf16_t* Ab = A  + (size_t)blockIdx.z * (size_t)sA;
    const bf16_t* Bb = Bt + (size_t)blockIdx.z * (size_t)sB;

    const unsigned int asLds = (unsigned int)(size_t)(void*)&As[0];
    const unsigned int bsLds = (unsigned int)(size_t)(void*)&Bs[0];

    v8f acc[2][2] = {};

    // issue one stage (64 K-cols): TDM for A, async-to-LDS for B
    auto issue = [&](int k0, int bufsel) {
        if (wave == 0) {
            unsigned int abase = asLds + (unsigned)bufsel * (AROWS * LPAD * 2);
            tdm_load_tile(abase, Ab + (size_t)m0 * lda + k0, AROWS, lda);
        }
        unsigned int bbase = bsLds + (unsigned)bufsel * (BROWS * LPAD * 2);
#pragma unroll
        for (int i = 0; i < 2; ++i) {              // B: 64x64 = 512 chunks
            int c   = tid + i * 256;
            int row = c >> 3;
            int col = (c & 7) << 3;
            const bf16_t* src = Bb + (size_t)(n0 + row) * ldb + k0 + col;
            unsigned int dst = bbase + (unsigned)(row * LPAD + col) * 2;
            asm volatile("global_load_async_to_lds_b128 %0, %1, off"
                         :: "v"(dst), "v"(src) : "memory");
        }
    };

    // consume one staged buffer: 2 K-steps x (2x2) tiles = 8 WMMAs
    auto compute = [&](int cur) {
        const bf16_t* as = &As[(size_t)cur * (AROWS * LPAD)];
        const bf16_t* bs = &Bs[(size_t)cur * (BROWS * LPAD)];
#pragma unroll
        for (int kk = 0; kk < 64; kk += 32) {
            v16bf af[2];
#pragma unroll
            for (int mt = 0; mt < 2; ++mt) {
                const bf16_t* ar =
                    &as[(wm * 32 + mt * 16 + lr) * LPAD + kk + 8 * half];
                v8bf a0 = *(const v8bf*)ar;
                v8bf a1 = *(const v8bf*)(ar + 16);
#pragma unroll
                for (int i = 0; i < 8; ++i) {
                    af[mt][i] = a0[i]; af[mt][i + 8] = a1[i];
                }
            }
#pragma unroll
            for (int nt = 0; nt < 2; ++nt) {
                const bf16_t* br =
                    &bs[(wn * 32 + nt * 16 + lr) * LPAD + kk + 8 * half];
                v8bf b0 = *(const v8bf*)br;
                v8bf b1 = *(const v8bf*)(br + 16);
                v16bf bfv;
#pragma unroll
                for (int i = 0; i < 8; ++i) { bfv[i] = b0[i]; bfv[i + 8] = b1[i]; }
                acc[0][nt] = __builtin_amdgcn_wmma_f32_16x16x32_bf16(
                    false, af[0], false, bfv, (short)0, acc[0][nt], false, false);
                acc[1][nt] = __builtin_amdgcn_wmma_f32_16x16x32_bf16(
                    false, af[1], false, bfv, (short)0, acc[1][nt], false, false);
            }
        }
    };

    const int nstages = K >> 6;       // K / 64
    issue(0, 0);

    // steady state: always one stage in flight
    for (int s = 0; s + 1 < nstages; ++s) {
        const int cur = s & 1;
        issue((s + 1) << 6, cur ^ 1);
        asm volatile("s_wait_asynccnt 2" ::: "memory");
        if (wave == 0) __builtin_amdgcn_s_wait_tensorcnt((short)1);
        __syncthreads();
        compute(cur);
        __syncthreads();
    }
    // peeled last stage
    asm volatile("s_wait_asynccnt 0" ::: "memory");
    if (wave == 0) __builtin_amdgcn_s_wait_tensorcnt((short)0);
    __syncthreads();
    compute((nstages - 1) & 1);

    // ---- epilogue: D layout -> lane lr = col, VGPR j -> row j + 8*half ----
    const size_t cb = (size_t)blockIdx.z * (size_t)sC;
#pragma unroll
    for (int mt = 0; mt < 2; ++mt) {
#pragma unroll
        for (int nt = 0; nt < 2; ++nt) {
            int gn = n0 + wn * 32 + nt * 16 + lr;
            float bv = bias ? bias[gn] : 0.0f;
#pragma unroll
            for (int j = 0; j < 8; ++j) {
                int gm = m0 + wm * 32 + mt * 16 + 8 * half + j;
                float v = acc[mt][nt][j] * scale + bv;
                if (actGelu) v = gelu_f(v);
                size_t idx = cb + (size_t)gm * ldc + gn;
                if (resid) v += resid[idx];
                if (outF) outF[idx] = v;
                if (outB) {
                    if (transOutB) {
                        // write V^T: [b][n][s] with b = gm/S_, s = gm%S_
                        size_t ti = ((size_t)(gm >> 11) * H_ + gn) * S_ +
                                    (size_t)(gm & (S_ - 1));
                        outB[ti] = (bf16_t)v;
                    } else {
                        outB[idx] = (bf16_t)v;
                    }
                }
            }
        }
    }
}

// ---------------------------------------------------------------------------
// float (K x N) -> bf16 transposed (N x K)
// ---------------------------------------------------------------------------
__global__ void f2bT_kernel(const float* __restrict__ x, bf16_t* __restrict__ y,
                            int Kdim, int Ndim)
{
    size_t i = (size_t)blockIdx.x * 256 + threadIdx.x;
    if (i >= (size_t)Kdim * Ndim) return;
    int k = (int)(i / Ndim);
    int n = (int)(i % Ndim);
    y[(size_t)n * Kdim + k] = (bf16_t)x[i];
}

// ---------------------------------------------------------------------------
// Masked mean pool over sequence: pooled[b][h]
// ---------------------------------------------------------------------------
__global__ void pool_kernel(const float* __restrict__ hs,
                            const int* __restrict__ mask,
                            float* __restrict__ pooled)
{
    int b = blockIdx.x;
    int h = blockIdx.y * 256 + threadIdx.x;
    float s = 0.0f, cnt = 0.0f;
    for (int ss = 0; ss < S_; ++ss) {
        float m = (float)mask[b * S_ + ss];
        s   += m * hs[((size_t)b * S_ + ss) * H_ + h];
        cnt += m;
    }
    pooled[b * H_ + h] = s / (cnt + 1e-8f);
}

// ---------------------------------------------------------------------------
// Small dense:  out[b][o] = act(in[b] . W[:,o] + bias[o])   (W row-major KxN)
// ---------------------------------------------------------------------------
__global__ void gemv_small(const float* __restrict__ in,
                           const float* __restrict__ W,
                           const float* __restrict__ bias,
                           float* __restrict__ out, int K, int N, int act)
{
    int b = blockIdx.x;
    int o = blockIdx.y * 256 + threadIdx.x;
    if (o >= N) return;
    const float* x = in + (size_t)b * K;
    float s = bias[o];
    for (int k = 0; k < K; ++k) s += x[k] * W[(size_t)k * N + o];
    if (act) s = gelu_f(s);
    out[(size_t)b * N + o] = s;
}

// ---------------------------------------------------------------------------
// Row layer-norm, width W (one block per row)
// ---------------------------------------------------------------------------
__global__ void __launch_bounds__(256) layernorm_kernel(
    const float* __restrict__ x, const float* __restrict__ g,
    const float* __restrict__ bb, float* __restrict__ out, int W)
{
    __shared__ float red[256];
    int row = blockIdx.x, tid = threadIdx.x;
    const float* xr = x + (size_t)row * W;
    float s = 0.0f;
    for (int i = tid; i < W; i += 256) s += xr[i];
    red[tid] = s; __syncthreads();
    for (int st = 128; st > 0; st >>= 1) {
        if (tid < st) red[tid] += red[tid + st];
        __syncthreads();
    }
    float mean = red[0] / (float)W;
    __syncthreads();
    float v = 0.0f;
    for (int i = tid; i < W; i += 256) { float d = xr[i] - mean; v += d * d; }
    red[tid] = v; __syncthreads();
    for (int st = 128; st > 0; st >>= 1) {
        if (tid < st) red[tid] += red[tid + st];
        __syncthreads();
    }
    float inv = rsqrtf(red[0] / (float)W + EPS_);
    for (int i = tid; i < W; i += 256)
        out[(size_t)row * W + i] = (xr[i] - mean) * inv * g[i] + bb[i];
}

// ---------------------------------------------------------------------------
// Gumbel straight-through: forward value == one-hot(argmax(logits+noise)),
// so z_codes[b][e] = codebook[argmax].
// ---------------------------------------------------------------------------
__global__ void gumbel_code_kernel(const float* __restrict__ zraw,
                                   const float* __restrict__ gnoise,
                                   const float* __restrict__ codebook,
                                   float* __restrict__ codes)
{
    int t = threadIdx.x;           // one thread per (b,e)
    if (t >= B_ * NE_) return;
    int b = t / NE_, e = t % NE_;
    const float* z = zraw  + (size_t)b * (NE_ * NC_) + e * NC_;
    const float* g = gnoise + (size_t)(b * NE_ + e) * NC_;
    int best = 0; float bv = z[0] + g[0];
    for (int c = 1; c < NC_; ++c) {
        float v = z[c] + g[c];
        if (v > bv) { bv = v; best = c; }
    }
    float* dst = codes + (size_t)(b * NE_ + e) * CD_;
    const float* src = codebook + (size_t)best * CD_;
    for (int d = 0; d < CD_; ++d) dst[d] = src[d];
}

// ---------------------------------------------------------------------------
// Per-(b,e) world-knowledge MLP: concat(h_ctx, code) -> 2048 (gelu) -> 128
// ---------------------------------------------------------------------------
__global__ void __launch_bounds__(256) wk_kernel(
    const float* __restrict__ hctx, const float* __restrict__ codes,
    const float* __restrict__ w1, const float* __restrict__ b1,
    const float* __restrict__ w2, const float* __restrict__ b2,
    float* __restrict__ outcome)
{
    __shared__ float hid[2048];
    int be = blockIdx.x;
    int b  = be / NE_;
    int tid = threadIdx.x;
    const float* hx = hctx + (size_t)b * H_;
    const float* cd = codes + (size_t)be * CD_;
#pragma unroll
    for (int i = 0; i < 8; ++i) {
        int o = tid + i * 256;
        float s = b1[o];
        for (int k = 0; k < H_; ++k)  s += hx[k] * w1[(size_t)k * 2048 + o];
        for (int k = 0; k < CD_; ++k) s += cd[k] * w1[(size_t)(H_ + k) * 2048 + o];
        hid[o] = gelu_f(s);
    }
    __syncthreads();
    if (tid < OD_) {
        float s = b2[tid];
        for (int k = 0; k < 2048; ++k) s += hid[k] * w2[(size_t)k * OD_ + tid];
        outcome[(size_t)be * OD_ + tid] = s;
    }
}

// ---------------------------------------------------------------------------
// Target projection + info-gain selection + FiLM generator -> gamma/beta
// ---------------------------------------------------------------------------
__global__ void __launch_bounds__(256) select_film_kernel(
    const float* __restrict__ hctx, const float* __restrict__ outcome,
    const float* __restrict__ tpw, const float* __restrict__ filmw,
    const float* __restrict__ filmb,
    float* __restrict__ gamma, float* __restrict__ beta)
{
    __shared__ float tv[OD_];
    __shared__ float red[256];
    __shared__ float ig[NE_];
    __shared__ int   sidx;
    int b = blockIdx.x, tid = threadIdx.x;
    const float* hx = hctx + (size_t)b * H_;

    if (tid < OD_) {
        float s = 0.0f;
        for (int k = 0; k < H_; ++k) s += hx[k] * tpw[(size_t)k * OD_ + tid];
        tv[tid] = s;
    }
    __syncthreads();

    red[tid] = (tid < OD_) ? tv[tid] * tv[tid] : 0.0f;
    __syncthreads();
    for (int st = 128; st > 0; st >>= 1) {
        if (tid < st) red[tid] += red[tid + st];
        __syncthreads();
    }
    float tnorm = fmaxf(sqrtf(red[0]), 1e-12f);
    __syncthreads();

    for (int e = 0; e < NE_; ++e) {
        const float* oc = outcome + (size_t)(b * NE_ + e) * OD_;
        red[tid] = (tid < OD_) ? oc[tid] * oc[tid] : 0.0f;
        __syncthreads();
        for (int st = 128; st > 0; st >>= 1) {
            if (tid < st) red[tid] += red[tid + st];
            __syncthreads();
        }
        float onorm = fmaxf(sqrtf(red[0]), 1e-12f);
        __syncthreads();
        red[tid] = (tid < OD_) ? (oc[tid] * tv[tid]) / (onorm * tnorm) : 0.0f;
        __syncthreads();
        for (int st = 128; st > 0; st >>= 1) {
            if (tid < st) red[tid] += red[tid + st];
            __syncthreads();
        }
        if (tid == 0) ig[e] = red[0] / TEMP_;
        __syncthreads();
    }
    if (tid == 0) {
        int bi = 0; float bv = ig[0];
        for (int e = 1; e < NE_; ++e)
            if (ig[e] > bv) { bv = ig[e]; bi = e; }
        sidx = bi;
    }
    __syncthreads();
    const float* chosen = outcome + (size_t)(b * NE_ + sidx) * OD_;
#pragma unroll
    for (int i = 0; i < 8; ++i) {
        int o = tid + i * 256;
        float s = filmb[o];
        for (int k = 0; k < OD_; ++k) s += chosen[k] * filmw[(size_t)k * 2048 + o];
        s = gelu_f(s);
        if (o < H_) gamma[(size_t)b * H_ + o] = s;
        else        beta[(size_t)b * H_ + (o - H_)] = s;
    }
}

// ---------------------------------------------------------------------------
// FiLM fuse: fused = gamma[b]*hs + beta[b]   (writes f32 + bf16)
// ---------------------------------------------------------------------------
__global__ void film_fuse_kernel(const float* __restrict__ hs,
                                 const float* __restrict__ gamma,
                                 const float* __restrict__ beta,
                                 float* __restrict__ fF,
                                 bf16_t* __restrict__ fB)
{
    size_t i = (size_t)blockIdx.x * 256 + threadIdx.x;
    int h = (int)(i % H_);
    int b = (int)(i / ((size_t)S_ * H_));
    float v = gamma[b * H_ + h] * hs[i] + beta[b * H_ + h];
    fF[i] = v;
    fB[i] = (bf16_t)v;
}

// ---------------------------------------------------------------------------
// Masked row softmax over S columns; emits bf16 probabilities
// ---------------------------------------------------------------------------
__global__ void __launch_bounds__(256) softmax_kernel(
    const float* __restrict__ scores, const int* __restrict__ mask,
    bf16_t* __restrict__ P)
{
    __shared__ float red[256];
    int row = blockIdx.x;
    int b   = row / S_;
    int tid = threadIdx.x;
    const float* sr = scores + (size_t)row * S_;
    const int*   mr = mask + (size_t)b * S_;

    float mx = -3.4e38f;
    for (int k = tid; k < S_; k += 256)
        if (mr[k] != 0) mx = fmaxf(mx, sr[k]);
    red[tid] = mx; __syncthreads();
    for (int st = 128; st > 0; st >>= 1) {
        if (tid < st) red[tid] = fmaxf(red[tid], red[tid + st]);
        __syncthreads();
    }
    mx = red[0]; __syncthreads();

    float s = 0.0f;
    for (int k = tid; k < S_; k += 256)
        if (mr[k] != 0) s += __expf(sr[k] - mx);
    red[tid] = s; __syncthreads();
    for (int st = 128; st > 0; st >>= 1) {
        if (tid < st) red[tid] += red[tid + st];
        __syncthreads();
    }
    float inv = 1.0f / red[0];

    for (int k = tid; k < S_; k += 256) {
        float v = (mr[k] != 0) ? __expf(sr[k] - mx) * inv : 0.0f;
        P[(size_t)row * S_ + k] = (bf16_t)v;
    }
}

// ---------------------------------------------------------------------------
// Host-side orchestration
// ---------------------------------------------------------------------------
extern "C" void kernel_launch(void* const* d_in, const int* in_sizes, int n_in,
                              void* d_out, int out_size, void* d_ws,
                              size_t ws_size, hipStream_t stream)
{
    const float* hs       = (const float*)d_in[0];
    const int*   mask     = (const int*)  d_in[1];
    const float* gnoise   = (const float*)d_in[2];
    const float* cp_w     = (const float*)d_in[3];
    const float* cp_b     = (const float*)d_in[4];
    const float* cp_ln_g  = (const float*)d_in[5];
    const float* cp_ln_b  = (const float*)d_in[6];
    const float* eph_w1   = (const float*)d_in[7];
    const float* eph_b1   = (const float*)d_in[8];
    const float* eph_w2   = (const float*)d_in[9];
    const float* eph_b2   = (const float*)d_in[10];
    const float* codebook = (const float*)d_in[11];
    const float* wk_w1    = (const float*)d_in[12];
    const float* wk_b1    = (const float*)d_in[13];
    const float* wk_w2    = (const float*)d_in[14];
    const float* wk_b2    = (const float*)d_in[15];
    const float* tp_w     = (const float*)d_in[16];
    const float* film_w   = (const float*)d_in[17];
    const float* film_b   = (const float*)d_in[18];
    const float* q_w      = (const float*)d_in[19];
    const float* q_b      = (const float*)d_in[20];
    const float* k_w      = (const float*)d_in[21];
    const float* k_b      = (const float*)d_in[22];
    const float* v_w      = (const float*)d_in[23];
    const float* v_b      = (const float*)d_in[24];
    const float* o_w      = (const float*)d_in[25];
    const float* o_b      = (const float*)d_in[26];
    const float* int_w    = (const float*)d_in[27];
    const float* int_b    = (const float*)d_in[28];
    const float* out_w    = (const float*)d_in[29];
    const float* out_b    = (const float*)d_in[30];
    const float* ln_g     = (const float*)d_in[31];
    const float* ln_b     = (const float*)d_in[32];

    char* base = (char*)d_ws;
    size_t off = 0;
    auto alloc = [&](size_t bytes) -> void* {
        void* p = base + off;
        off = (off + bytes + 255) & ~(size_t)255;
        return p;
    };

    const size_t BSH = (size_t)B_ * S_ * H_;
    const size_t BSS = (size_t)B_ * S_ * S_;

    float*  fusedF = (float*) alloc(BSH * 4);
    bf16_t* fusedB = (bf16_t*)alloc(BSH * 2);
    bf16_t* qb     = (bf16_t*)alloc(BSH * 2);          // reused as attn_out bf16
    bf16_t* kb     = (bf16_t*)alloc(BSH * 2);          // N x K layout already
    bf16_t* vbT    = (bf16_t*)alloc(BSH * 2);          // V^T: [b][H][S]
    float*  scores = (float*) alloc(BSS * 4);          // reused as FFN hidden bf16
    bf16_t* Pb     = (bf16_t*)alloc(BSS * 2);          // reused as pre-LN f32
    float*  x1F    = (float*) alloc(BSH * 4);
    bf16_t* x1B    = (bf16_t*)alloc(BSH * 2);
    bf16_t* wqT    = (bf16_t*)alloc((size_t)H_ * H_ * 2);
    bf16_t* wkT    = (bf16_t*)alloc((size_t)H_ * H_ * 2);
    bf16_t* wvT    = (bf16_t*)alloc((size_t)H_ * H_ * 2);
    bf16_t* woT    = (bf16_t*)alloc((size_t)H_ * H_ * 2);
    bf16_t* wintT  = (bf16_t*)alloc((size_t)H_ * 4 * H_ * 2);
    bf16_t* woutT  = (bf16_t*)alloc((size_t)4 * H_ * H_ * 2);
    float* pooled  = (float*)alloc((size_t)B_ * H_ * 4);
    float* tmp1    = (float*)alloc((size_t)B_ * H_ * 4);
    float* hctx    = (float*)alloc((size_t)B_ * H_ * 4);
    float* z1      = (float*)alloc((size_t)B_ * H_ * 4);
    float* zraw    = (float*)alloc((size_t)B_ * NE_ * NC_ * 4);
    float* codes   = (float*)alloc((size_t)B_ * NE_ * CD_ * 4);
    float* outcome = (float*)alloc((size_t)B_ * NE_ * OD_ * 4);
    float* gamma   = (float*)alloc((size_t)B_ * H_ * 4);
    float* beta    = (float*)alloc((size_t)B_ * H_ * 4);

    bf16_t* attnb = qb;               // alias: qb dead after scores GEMM
    bf16_t* h4b   = (bf16_t*)scores;  // alias: scores dead after softmax
    float*  yF    = (float*)Pb;       // alias: P dead after attn@V

    // ---- weight conversion: f32 (KxN) -> bf16 transposed (NxK) ----
    const size_t nHH = (size_t)H_ * H_;
    f2bT_kernel<<<(unsigned)((nHH + 255) / 256), 256, 0, stream>>>(q_w, wqT, H_, H_);
    f2bT_kernel<<<(unsigned)((nHH + 255) / 256), 256, 0, stream>>>(k_w, wkT, H_, H_);
    f2bT_kernel<<<(unsigned)((nHH + 255) / 256), 256, 0, stream>>>(v_w, wvT, H_, H_);
    f2bT_kernel<<<(unsigned)((nHH + 255) / 256), 256, 0, stream>>>(o_w, woT, H_, H_);
    f2bT_kernel<<<(unsigned)((nHH * 4 + 255) / 256), 256, 0, stream>>>(int_w, wintT, H_, 4 * H_);
    f2bT_kernel<<<(unsigned)((nHH * 4 + 255) / 256), 256, 0, stream>>>(out_w, woutT, 4 * H_, H_);

    // ---- tiny head phase ----
    pool_kernel<<<dim3(B_, H_ / 256), 256, 0, stream>>>(hs, mask, pooled);
    gemv_small<<<dim3(B_, (H_ + 255) / 256), 256, 0, stream>>>(
        pooled, cp_w, cp_b, tmp1, H_, H_, 1);
    layernorm_kernel<<<B_, 256, 0, stream>>>(tmp1, cp_ln_g, cp_ln_b, hctx, H_);
    gemv_small<<<dim3(B_, (H_ + 255) / 256), 256, 0, stream>>>(
        hctx, eph_w1, eph_b1, z1, H_, H_, 1);
    gemv_small<<<dim3(B_, 1), 256, 0, stream>>>(
        z1, eph_w2, eph_b2, zraw, H_, NE_ * NC_, 0);
    gumbel_code_kernel<<<1, 32, 0, stream>>>(zraw, gnoise, codebook, codes);
    wk_kernel<<<B_ * NE_, 256, 0, stream>>>(
        hctx, codes, wk_w1, wk_b1, wk_w2, wk_b2, outcome);
    select_film_kernel<<<B_, 256, 0, stream>>>(
        hctx, outcome, tp_w, film_w, film_b, gamma, beta);
    film_fuse_kernel<<<(unsigned)(BSH / 256), 256, 0, stream>>>(
        hs, gamma, beta, fusedF, fusedB);

    // ---- heavy phase: all GEMMs via WMMA (B operands all N x K) ----
    const int M = B_ * S_;
    dim3 g1(H_ / 64, M / 128, 1);
    // Q, K, V projections (V writes V^T directly)
    gemm_wmma<<<g1, 256, 0, stream>>>(fusedB, wqT, q_b, nullptr, nullptr, qb,
        M, H_, H_, H_, H_, H_, 0, 0, 0, 1.0f, 0, 0);
    gemm_wmma<<<g1, 256, 0, stream>>>(fusedB, wkT, k_b, nullptr, nullptr, kb,
        M, H_, H_, H_, H_, H_, 0, 0, 0, 1.0f, 0, 0);
    gemm_wmma<<<g1, 256, 0, stream>>>(fusedB, wvT, v_b, nullptr, nullptr, vbT,
        M, H_, H_, H_, H_, H_, 0, 0, 0, 1.0f, 0, 1);

    // scores = Q @ K^T / sqrt(H)   (batched; K rows are already N x K)
    dim3 gs(S_ / 64, S_ / 128, B_);
    gemm_wmma<<<gs, 256, 0, stream>>>(qb, kb, nullptr, nullptr, scores, nullptr,
        S_, S_, H_, H_, H_, S_,
        (long long)S_ * H_, (long long)S_ * H_, (long long)S_ * S_,
        0.03125f /* 1/sqrt(1024) */, 0, 0);

    // masked softmax -> bf16 P
    softmax_kernel<<<B_ * S_, 256, 0, stream>>>(scores, mask, Pb);

    // attn = P @ V = P @ (V^T)^T  (batched, B operand = V^T [H][S])
    dim3 ga(H_ / 64, S_ / 128, B_);
    gemm_wmma<<<ga, 256, 0, stream>>>(Pb, vbT, nullptr, nullptr, nullptr, attnb,
        S_, H_, S_, S_, S_, H_,
        (long long)S_ * S_, (long long)H_ * S_, (long long)S_ * H_,
        1.0f, 0, 0);

    // o-projection + bias + residual(fused) -> x1 (f32 + bf16)
    gemm_wmma<<<g1, 256, 0, stream>>>(attnb, woT, o_b, fusedF, x1F, x1B,
        M, H_, H_, H_, H_, H_, 0, 0, 0, 1.0f, 0, 0);

    // FFN up (gelu) -> h4 bf16
    dim3 g4(4 * H_ / 64, M / 128, 1);
    gemm_wmma<<<g4, 256, 0, stream>>>(x1B, wintT, int_b, nullptr, nullptr, h4b,
        M, 4 * H_, H_, H_, H_, 4 * H_, 0, 0, 0, 1.0f, 1, 0);

    // FFN down + bias + residual(x1) -> yF
    gemm_wmma<<<g1, 256, 0, stream>>>(h4b, woutT, out_b, x1F, yF, nullptr,
        M, H_, 4 * H_, 4 * H_, 4 * H_, H_, 0, 0, 0, 1.0f, 0, 0);

    // final layer-norm -> d_out (f32)
    layernorm_kernel<<<B_ * S_, 256, 0, stream>>>(
        yF, ln_g, ln_b, (float*)d_out, H_);
}
